// MESHEncoder_4037269258856
// MI455X (gfx1250) — matrix operand order repaired and checked by reading, not compiled, not run on hardware
//
#include <hip/hip_runtime.h>
#include <math.h>

typedef __attribute__((ext_vector_type(16))) _Float16 v16h;
typedef __attribute__((ext_vector_type(8)))  _Float16 v8h;
typedef __attribute__((ext_vector_type(8)))  float    v8f;

#define BB 4
#define SS 2048
#define DD 1024
#define KK 128
#define MROWS (BB*SS)
#define EPSV 0.05f
#define INV_EPS 20.0f
#define NITERS 50
#define LOG_A (-7.6246189861593985f)    /* -ln(2048) */
#define LOG_B (-4.8520302639196169f)    /* -ln(128)  */
#define PHC   (-0.0089944730195079920f) /* -ln(10000)/1024 */
#define NEG_INF (-__builtin_inff())

// ---------------- wave32 helpers ----------------
__device__ __forceinline__ float wredf(float v){
#pragma unroll
  for (int m = 16; m > 0; m >>= 1) v += __shfl_xor(v, m, 32);
  return v;
}
__device__ __forceinline__ int wredi(int v){
#pragma unroll
  for (int m = 16; m > 0; m >>= 1) v += __shfl_xor(v, m, 32);
  return v;
}
__device__ __forceinline__ void lse_acc(float& m, float& s, float v){
  if (v > m){ s = s * __expf(m - v) + 1.0f; m = v; }
  else      { s += __expf(v - m); }
}
__device__ __forceinline__ void lse_merge(float& m, float& s, float m2, float s2){
  float M = fmaxf(m, m2);
  float S = ((m  == NEG_INF) ? 0.0f : s  * __expf(m  - M)) +
            ((m2 == NEG_INF) ? 0.0f : s2 * __expf(m2 - M));
  m = M; s = S;
}
__device__ __forceinline__ void lse_wred(float& m, float& s){
#pragma unroll
  for (int d = 16; d > 0; d >>= 1){
    float m2 = __shfl_xor(m, d, 32);
    float s2 = __shfl_xor(s, d, 32);
    lse_merge(m, s, m2, s2);
  }
}

// ---------------- K0: transpose + f16-convert weights ----------------
__global__ void __launch_bounds__(256)
prep_kernel(const float* __restrict__ Wc, const float* __restrict__ Wo,
            _Float16* __restrict__ Bt, _Float16* __restrict__ Wot){
  int i = blockIdx.x * blockDim.x + threadIdx.x;
  if (i < DD * KK){                 // W_cost (D,K) -> Bt[k][d]
    int d = i >> 7, k = i & 127;
    Bt[(size_t)k * DD + d] = (_Float16)Wc[i];
  } else {                          // W_out (K,D) -> Wot[d][k]
    int j = i - DD * KK;
    int k = j >> 10, d = j & 1023;
    Wot[(size_t)d * KK + k] = (_Float16)Wo[j];
  }
}

// ---------------- K1: C = gather(emb) @ W_cost + b_cost (WMMA) ----------------
__global__ void __launch_bounds__(256)
cost_gemm_kernel(const int* __restrict__ tok, const float* __restrict__ emb,
                 const _Float16* __restrict__ Bt, const float* __restrict__ bcost,
                 float* __restrict__ C){
  int wid  = threadIdx.x >> 5;
  int lane = threadIdx.x & 31;
  int tile = blockIdx.x * 8 + wid;     // 4096 tiles = 512 mTiles x 8 kTiles
  int mT = tile >> 3, kT = tile & 7;
  int row0 = mT << 4, k0 = kT << 4;
  int n  = lane & 15;
  int hi = lane >> 4;

  const float*    arow = emb + (size_t)tok[row0 + n] * DD;   // row M = lane&15
  const _Float16* bp   = Bt + (size_t)(k0 + n) * DD + hi * 16;

  v8f acc = {};
#pragma unroll 2
  for (int d = 0; d < DD; d += 32){
    int c0 = d + hi * 16;
    float4 p0 = *(const float4*)(arow + c0);
    float4 p1 = *(const float4*)(arow + c0 + 4);
    float4 p2 = *(const float4*)(arow + c0 + 8);
    float4 p3 = *(const float4*)(arow + c0 + 12);
    v16h a;
    a[0]=(_Float16)p0.x;  a[1]=(_Float16)p0.y;  a[2]=(_Float16)p0.z;  a[3]=(_Float16)p0.w;
    a[4]=(_Float16)p1.x;  a[5]=(_Float16)p1.y;  a[6]=(_Float16)p1.z;  a[7]=(_Float16)p1.w;
    a[8]=(_Float16)p2.x;  a[9]=(_Float16)p2.y;  a[10]=(_Float16)p2.z; a[11]=(_Float16)p2.w;
    a[12]=(_Float16)p3.x; a[13]=(_Float16)p3.y; a[14]=(_Float16)p3.z; a[15]=(_Float16)p3.w;
    v16h b = *(const v16h*)(bp + d);   // 32B contiguous: K = hi*16 + 0..15
    acc = __builtin_amdgcn_wmma_f32_16x16x32_f16(false, a, false, b, (short)0, acc, false, false);
  }
  int col = k0 + n;
  float bias = bcost[col];
#pragma unroll
  for (int r = 0; r < 8; ++r){
    int row = row0 + r + hi * 8;
    C[(size_t)row * KK + col] = acc[r] + bias;
  }
}

// ---------------- K2: log-domain Sinkhorn, one workgroup per batch ----------------
__global__ void __launch_bounds__(1024)
sinkhorn_kernel(const float* __restrict__ C, float* __restrict__ fOut, float* __restrict__ gOut){
  __shared__ float fL[SS];
  __shared__ float gL[KK];
  __shared__ float pm[8 * KK];
  __shared__ float ps[8 * KK];
  int b = blockIdx.x;
  int t = threadIdx.x;
  int lane = t & 31;
  int wv   = t >> 5;
  const float* Cb = C + (size_t)b * SS * KK;
  if (t < KK) gL[t] = 0.0f;
  __syncthreads();

  for (int it = 0; it < NITERS; ++it){
    // f-update: one wave per row, lane covers k = 4*lane..4*lane+3, 64 rows/wave
    float4 gv = *(const float4*)&gL[lane * 4];
    for (int r = 0; r < 64; ++r){
      int s = wv * 64 + r;
      float4 cv = *(const float4*)(Cb + (size_t)s * KK + lane * 4);
      float m = NEG_INF, sm = 0.0f;
      lse_acc(m, sm, (gv.x - cv.x) * INV_EPS);
      lse_acc(m, sm, (gv.y - cv.y) * INV_EPS);
      lse_acc(m, sm, (gv.z - cv.z) * INV_EPS);
      lse_acc(m, sm, (gv.w - cv.w) * INV_EPS);
      lse_wred(m, sm);
      if (lane == 0) fL[s] = EPSV * (LOG_A - (m + __logf(sm)));
    }
    __syncthreads();
    // g-update phase 1: thread -> (k = t&127, chunk = t>>7 of 256 rows), 4 indep chains
    {
      int k = t & 127, ch = t >> 7;
      const float* cc = Cb + (size_t)(ch * 256) * KK + k;
      float m0 = NEG_INF, s0 = 0.0f, m1 = NEG_INF, s1 = 0.0f;
      float m2 = NEG_INF, s2 = 0.0f, m3 = NEG_INF, s3 = 0.0f;
      int sb = ch * 256;
      for (int r = 0; r < 64; ++r){
        lse_acc(m0, s0, (fL[sb + r      ] - cc[(size_t)(r      ) * KK]) * INV_EPS);
        lse_acc(m1, s1, (fL[sb + r +  64] - cc[(size_t)(r +  64) * KK]) * INV_EPS);
        lse_acc(m2, s2, (fL[sb + r + 128] - cc[(size_t)(r + 128) * KK]) * INV_EPS);
        lse_acc(m3, s3, (fL[sb + r + 192] - cc[(size_t)(r + 192) * KK]) * INV_EPS);
      }
      lse_merge(m0, s0, m1, s1);
      lse_merge(m2, s2, m3, s3);
      lse_merge(m0, s0, m2, s2);
      pm[ch * KK + k] = m0; ps[ch * KK + k] = s0;
    }
    __syncthreads();
    if (t < KK){
      float M = NEG_INF, S = 0.0f;
#pragma unroll
      for (int c2 = 0; c2 < 8; ++c2) lse_merge(M, S, pm[c2 * KK + t], ps[c2 * KK + t]);
      gL[t] = EPSV * (LOG_B - (M + __logf(S)));
    }
    __syncthreads();
  }
  fOut[b * SS + t]        = fL[t];
  fOut[b * SS + t + 1024] = fL[t + 1024];
  if (t < KK) gOut[b * KK + t] = gL[t];
}

// ---------------- K3: T, exact 32nd order statistic, sparsify -> f16 (x2048) ------
__global__ void __launch_bounds__(256)
sparsify_kernel(const float* __restrict__ C, const float* __restrict__ fG,
                const float* __restrict__ gG, _Float16* __restrict__ Ts){
  int lane = threadIdx.x & 31;
  int row  = blockIdx.x * 8 + (threadIdx.x >> 5);
  int b = row >> 11;
  float fv = fG[row];
  const float* crow = C + (size_t)row * KK;
  const float* grow = gG + b * KK;

  float tv[4]; unsigned bits[4];
#pragma unroll
  for (int j = 0; j < 4; ++j){
    int k = lane + 32 * j;
    tv[j] = __expf((fv + grow[k] - crow[k]) * INV_EPS);  // T > 0 always
    bits[j] = __float_as_uint(tv[j]);
  }
  // radix-select the 32nd largest (positive floats are monotone as uints)
  unsigned cur = 0;
  for (int bit = 30; bit >= 0; --bit){
    unsigned test = cur | (1u << bit);
    int c = 0;
#pragma unroll
    for (int j = 0; j < 4; ++j) c += (bits[j] >= test);
    c = wredi(c);
    if (c >= 32) cur = test;
  }
  float tau = __uint_as_float(cur);

  int cg = 0;
#pragma unroll
  for (int j = 0; j < 4; ++j) cg += (tv[j] > tau);
  cg = wredi(cg);
  int need = 32 - cg;

  // tie rank in index order k = lane + 32*j (chunk-major, lane-minor)
  int eq[4], rank[4], base = 0;
#pragma unroll
  for (int j = 0; j < 4; ++j){
    eq[j] = (bits[j] == cur) ? 1 : 0;
    int incl = eq[j];
    for (int d = 1; d < 32; d <<= 1){
      int n2 = __shfl_up(incl, d, 32);
      if (lane >= d) incl += n2;
    }
    rank[j] = base + incl - eq[j];
    base += __shfl(incl, 31, 32);
  }

  float sp[4]; float sa = 0.0f;
#pragma unroll
  for (int j = 0; j < 4; ++j){ sp[j] = fmaxf(tv[j] - tau, 0.0f); sa += sp[j]; }
  sa = wredf(sa);
  if (sa <= 1e-12f){
#pragma unroll
    for (int j = 0; j < 4; ++j){
      bool msk = (tv[j] > tau) || (eq[j] && rank[j] < need);
      sp[j] = msk ? tv[j] : 0.0f;
    }
  }
#pragma unroll
  for (int j = 0; j < 4; ++j)
    Ts[(size_t)row * KK + lane + 32 * j] = (_Float16)(sp[j] * 2048.0f);
}

// ---------------- K4: sdr = T_sparse @ W_out + b_out, fused phase -> complex ------
__global__ void __launch_bounds__(256)
out_gemm_kernel(const _Float16* __restrict__ Ts, const _Float16* __restrict__ Wot,
                const float* __restrict__ bout, float2* __restrict__ out){
  int wid  = threadIdx.x >> 5;
  int lane = threadIdx.x & 31;
  int tile = blockIdx.x * 8 + wid;     // 32768 tiles = 512 mTiles x 64 nTiles
  int mT = tile >> 6, nT = tile & 63;
  int row0 = mT << 4, d0 = nT << 4;
  int n  = lane & 15;
  int hi = lane >> 4;

  const _Float16* pa = Ts  + (size_t)(row0 + n) * KK + hi * 16;
  const _Float16* pb = Wot + (size_t)(d0  + n) * KK + hi * 16;

  v8f acc = {};
#pragma unroll
  for (int kk = 0; kk < KK; kk += 32){
    v8h alo = *(const v8h*)(pa + kk);        // K = hi*16 + 0..7
    v8h ahi = *(const v8h*)(pa + kk + 8);    // K = hi*16 + 8..15
    v16h a;
#pragma unroll
    for (int i = 0; i < 8; ++i){ a[i] = alo[i]; a[8 + i] = ahi[i]; }
    v16h b = *(const v16h*)(pb + kk);        // K = hi*16 + 0..15
    acc = __builtin_amdgcn_wmma_f32_16x16x32_f16(false, a, false, b, (short)0, acc, false, false);
  }
  int col = d0 + n;
  float bias = bout[col];
  float divc = __expf((float)col * PHC);
#pragma unroll
  for (int r = 0; r < 8; ++r){
    int row = row0 + r + hi * 8;
    int s   = row & (SS - 1);
    float val = acc[r] * (1.0f / 2048.0f) + bias;   // undo T_sparse scaling
    float ph = (float)s * divc;
    float sn, cs;
    sincosf(ph, &sn, &cs);
    out[(size_t)row * DD + col] = make_float2(val * cs, val * sn);
  }
}

// ---------------- launch ----------------
extern "C" void kernel_launch(void* const* d_in, const int* in_sizes, int n_in,
                              void* d_out, int out_size, void* d_ws, size_t ws_size,
                              hipStream_t stream){
  (void)in_sizes; (void)n_in; (void)out_size; (void)ws_size;
  const int*   tok = (const int*)d_in[0];
  const float* emb = (const float*)d_in[1];
  const float* Wc  = (const float*)d_in[2];
  const float* bc  = (const float*)d_in[3];
  const float* Wo  = (const float*)d_in[4];
  const float* bo  = (const float*)d_in[5];

  char* ws = (char*)d_ws;                 // layout (bytes):
  float*    Cbuf = (float*)(ws + 0);          // 4,194,304  C (B,S,K) fp32
  float*    fG   = (float*)(ws + 4194304);    //    32,768  f (B,S)
  float*    gG   = (float*)(ws + 4227072);    //     2,048  g (B,K)
  _Float16* Ts   = (_Float16*)(ws + 4229120); // 2,097,152  T_sparse f16 (x2048)
  _Float16* Bt   = (_Float16*)(ws + 6326272); //   262,144  W_cost^T f16
  _Float16* Wot  = (_Float16*)(ws + 6588416); //   262,144  W_out^T  f16

  prep_kernel     <<<dim3(1024), dim3(256),  0, stream>>>(Wc, Wo, Bt, Wot);
  cost_gemm_kernel<<<dim3(512),  dim3(256),  0, stream>>>(tok, emb, Bt, bc, Cbuf);
  sinkhorn_kernel <<<dim3(4),    dim3(1024), 0, stream>>>(Cbuf, fG, gG);
  sparsify_kernel <<<dim3(1024), dim3(256),  0, stream>>>(Cbuf, fG, gG, Ts);
  out_gemm_kernel <<<dim3(4096), dim3(256),  0, stream>>>(Ts, Wot, bo, (float2*)d_out);
}
